// Attention_Module_32358283608447
// MI455X (gfx1250) — matrix operand order repaired and checked by reading, not compiled, not run on hardware
//
#include <hip/hip_runtime.h>
#include <hip/hip_bf16.h>

#define B_   2048
#define M_   64
#define D_   256
#define D2_  512
#define KREL 65536   // 256*256

typedef __attribute__((ext_vector_type(16))) __bf16 v16bf;
typedef __attribute__((ext_vector_type(8)))  float  v8f;

// ---------------------------------------------------------------- WMMA core
__device__ __forceinline__ v8f wmma_bf16(v16bf a, v16bf b, v8f c) {
  // D = A(16x32 bf16) * B(32x16 bf16) + C(16x16 f32)
  return __builtin_amdgcn_wmma_f32_16x16x32_bf16(
      /*neg_a=*/false, a, /*neg_b=*/false, b,
      /*c_mod=*/(short)0, c, /*reuse_a=*/false, /*reuse_b=*/false);
}

// A fragment from LDS bf16 tile (16 rows x 32 K), row stride ld elements.
// ISA 7.12.2 (16-bit A 16x32): lanes 0-15 row=lane, K {0..7,16..23};
// lanes 16-31 row=lane-16, K {8..15,24..31}.
__device__ __forceinline__ v16bf load_a_lds(const __bf16* base, int ld) {
  int lane = threadIdx.x & 31;
  int r  = lane & 15;
  int kb = (lane >> 4) * 8;
  const __bf16* p = base + r * ld + kb;
  v16bf a;
#pragma unroll
  for (int i = 0; i < 8; ++i) { a[i] = p[i]; a[8 + i] = p[16 + i]; }
  return a;
}

// B fragment from global bf16 in BT layout: BT[n*K + k] (column-n K-strip
// contiguous).  Lane n = lane&15; lanes 0-15 K {0..7,16..23}, lanes>=16
// K {8..15,24..31}.
__device__ __forceinline__ v16bf load_b_glb(const __bf16* bt, int K, int n0, int k0) {
  int lane = threadIdx.x & 31;
  int n = n0 + (lane & 15);
  int k = k0 + ((lane >> 4) * 8);
  const __bf16* p = bt + (size_t)n * K + k;
  v16bf b;
#pragma unroll
  for (int i = 0; i < 8; ++i) { b[i] = p[i]; b[8 + i] = p[16 + i]; }
  return b;
}

// C/D layout (32-bit 16x16): VGPR j -> row j (lanes 0-15) / row j+8 (16-31),
// col = lane&15.
__device__ __forceinline__ void store_c_f32(float* out, int ld, int row0, int col0, v8f c) {
  int lane = threadIdx.x & 31;
  int cn = col0 + (lane & 15);
  int rb = row0 + ((lane >> 4) * 8);
#pragma unroll
  for (int j = 0; j < 8; ++j) out[(size_t)(rb + j) * ld + cn] = c[j];
}

__device__ __forceinline__ float block_sum256(float x, float* tmp) {
#pragma unroll
  for (int off = 16; off; off >>= 1) x += __shfl_down(x, off);
  int wv = threadIdx.x >> 5, ln = threadIdx.x & 31;
  if (ln == 0) tmp[wv] = x;
  __syncthreads();
  float r = 0.f;
  if (threadIdx.x < 8) r = tmp[threadIdx.x];
  if (threadIdx.x < 32) {
#pragma unroll
    for (int off = 4; off; off >>= 1) r += __shfl_down(r, off);
    if (threadIdx.x == 0) tmp[0] = r;
  }
  __syncthreads();
  float out = tmp[0];
  __syncthreads();
  return out;
}

// ------------------------------------------------------------- small kernels
__global__ __launch_bounds__(256) void k_gather_pair(
    const int* __restrict__ pairs, const float* __restrict__ emb,
    float* __restrict__ eL, float* __restrict__ eR) {
  int b = blockIdx.x, d = threadIdx.x;
  int i0 = pairs[b * 2 + 0], i1 = pairs[b * 2 + 1];
  eL[(size_t)b * D_ + d] = emb[(size_t)i0 * D_ + d];
  eR[(size_t)b * D_ + d] = emb[(size_t)i1 * D_ + d];
}

// BT[n*K+k] = trans ? W[k*N+n] : W[n*K+k]   (bf16 convert)
__global__ __launch_bounds__(256) void k_cvt_bt(
    const float* __restrict__ w, __bf16* __restrict__ bt,
    int K, int N, int trans) {
  size_t i = (size_t)blockIdx.x * 256 + threadIdx.x;
  size_t tot = (size_t)K * N;
  if (i >= tot) return;
  size_t n = i / K, k = i - n * K;
  float v = trans ? w[k * (size_t)N + n] : w[n * (size_t)K + k];
  bt[i] = (__bf16)v;
}

// C(256x256) = op(A) @ B ;  ta=1 -> A used transposed.  (tiny precompute)
__global__ __launch_bounds__(256) void k_mm256(
    const float* __restrict__ A, const float* __restrict__ Bm,
    float* __restrict__ C, int ta) {
  int i = blockIdx.x, j = threadIdx.x;
  float acc = 0.f;
  for (int k = 0; k < 256; ++k) {
    float av = ta ? A[k * 256 + i] : A[i * 256 + k];
    acc = fmaf(av, Bm[k * 256 + j], acc);
  }
  C[i * 256 + j] = acc;
}

// --------------------------------------------------- bilinear rel_emb kernel
// rel_emb(2048 x 512) = P @ Wrel_bf^T, P[b, i*256+j] = eL[b,i]*eR[b,j].
// Block: 8 waves, 32 rows x 128 cols.  A frags built in-register from LDS.
__global__ __launch_bounds__(256) void k_bilinear(
    const float* __restrict__ eL, const float* __restrict__ eR,
    const __bf16* __restrict__ WrelB, float* __restrict__ relout) {
  __shared__ __bf16 sLb[32 * 256];
  __shared__ __bf16 sRb[32 * 256];
  int rg0 = blockIdx.x * 32;
  for (int e = threadIdx.x; e < 32 * 256; e += 256) {
    int r = e >> 8, d = e & 255;
    sLb[e] = (__bf16)eL[(size_t)(rg0 + r) * 256 + d];
    sRb[e] = (__bf16)eR[(size_t)(rg0 + r) * 256 + d];
  }
  __syncthreads();

  int wv = threadIdx.x >> 5;
  int lane = threadIdx.x & 31;
  int n0 = blockIdx.y * 128 + wv * 16;
  int rr0 = lane & 15, rr1 = rr0 + 16;
  int kb = (lane >> 4) * 8;
  v8f acc0 = {}, acc1 = {};

  for (int kc = 0; kc < 2048; ++kc) {
    int i = kc >> 3, j0 = (kc & 7) * 32;
    float l0 = (float)sLb[rr0 * 256 + i];
    float l1 = (float)sLb[rr1 * 256 + i];
    const __bf16* p0 = &sRb[rr0 * 256 + j0 + kb];
    const __bf16* p1 = &sRb[rr1 * 256 + j0 + kb];
    v16bf a0, a1;
#pragma unroll
    for (int t = 0; t < 8; ++t) {
      a0[t]     = (__bf16)(l0 * (float)p0[t]);
      a0[8 + t] = (__bf16)(l0 * (float)p0[16 + t]);
      a1[t]     = (__bf16)(l1 * (float)p1[t]);
      a1[8 + t] = (__bf16)(l1 * (float)p1[16 + t]);
    }
    v16bf bf = load_b_glb(WrelB, KREL, n0, kc * 32);
    // stream-ahead prefetch of the L2-resident Wrel (emits global_prefetch_b8)
    __builtin_prefetch(WrelB + (size_t)(n0 + (lane & 15)) * KREL + kc * 32 + 4096, 0, 0);
    acc0 = wmma_bf16(a0, bf, acc0);
    acc1 = wmma_bf16(a1, bf, acc1);
  }
  store_c_f32(relout, 512, rg0,      n0, acc0);
  store_c_f32(relout, 512, rg0 + 16, n0, acc1);
}

// --------------------------------------- fused gather + gate + relu (V matrix)
// rows = B*M ; V = relu(g*rel + (1-g)*ent), g = sigmoid(rel@W + ent@U + b)
__global__ __launch_bounds__(256) void k_gates(
    const int* __restrict__ conn, const float* __restrict__ emb,
    const __bf16* __restrict__ WT, const __bf16* __restrict__ UT,
    const float* __restrict__ bias, __bf16* __restrict__ Vout) {
  __shared__ float  relF[16 * 256];
  __shared__ float  entF[16 * 256];
  __shared__ __bf16 relB[16 * 264];
  __shared__ __bf16 entB[16 * 264];
  __shared__ int sidx[32];

  int rg0 = blockIdx.x * 16;
  if (threadIdx.x < 16) {
    int rg = rg0 + threadIdx.x;
    sidx[threadIdx.x]      = conn[rg * 2 + 0];
    sidx[16 + threadIdx.x] = conn[rg * 2 + 1];
  }
  __syncthreads();
  for (int e = threadIdx.x; e < 16 * 256; e += 256) {
    int r = e >> 8, d = e & 255;
    float rv = emb[(size_t)sidx[r] * 256 + d];
    float ev = emb[(size_t)sidx[16 + r] * 256 + d];
    relF[e] = rv; entF[e] = ev;
    relB[r * 264 + d] = (__bf16)rv;
    entB[r * 264 + d] = (__bf16)ev;
  }
  __syncthreads();

  int wv = threadIdx.x >> 5;
  int lane = threadIdx.x & 31;
  for (int nt = wv * 2; nt < wv * 2 + 2; ++nt) {
    int n0 = nt * 16;
    v8f acc = {};
    for (int kc = 0; kc < 8; ++kc) {
      int k0 = kc * 32;
      acc = wmma_bf16(load_a_lds(relB + k0, 264), load_b_glb(WT, 256, n0, k0), acc);
      acc = wmma_bf16(load_a_lds(entB + k0, 264), load_b_glb(UT, 256, n0, k0), acc);
    }
    int cn = n0 + (lane & 15);
    int rb = (lane >> 4) * 8;
#pragma unroll
    for (int j = 0; j < 8; ++j) {
      int r = rb + j;
      float pre  = acc[j] + bias[cn];
      float gsig = 1.f / (1.f + __expf(-pre));
      float vv = fmaxf(gsig * relF[r * 256 + cn] + (1.f - gsig) * entF[r * 256 + cn], 0.f);
      Vout[(size_t)(rg0 + r) * 256 + cn] = (__bf16)vv;
    }
  }
}

// ------------------------------------------- generic dual-input WMMA row GEMM
// C = act( A@BT (+ A2@BT2) + bias ) ; rows = gridDim.x*16, cols = gridDim.y*128
__global__ __launch_bounds__(256) void k_gemm(
    const float* __restrict__ A, int lda, const __bf16* __restrict__ BT,
    const float* __restrict__ A2, int lda2, const __bf16* __restrict__ BT2,
    const float* __restrict__ bias, int relu_act,
    float* __restrict__ C, int ldc, int K) {
  __shared__ __bf16 Ab [16 * 520];
  __shared__ __bf16 Ab2[16 * 520];
  int r0 = blockIdx.x * 16;
  int ldA = K + 8;
  for (int e = threadIdx.x; e < 16 * K; e += 256) {
    int r = e / K, k = e - r * K;
    Ab[r * ldA + k] = (__bf16)A[(size_t)(r0 + r) * lda + k];
    if (A2) Ab2[r * ldA + k] = (__bf16)A2[(size_t)(r0 + r) * lda2 + k];
  }
  __syncthreads();

  int wv = threadIdx.x >> 5;
  int lane = threadIdx.x & 31;
  int n0 = blockIdx.y * 128 + wv * 16;
  v8f acc = {};
  int nk = K >> 5;
  for (int kc = 0; kc < nk; ++kc) {
    int k0 = kc * 32;
    acc = wmma_bf16(load_a_lds(Ab + k0, ldA), load_b_glb(BT, K, n0, k0), acc);
    if (A2)
      acc = wmma_bf16(load_a_lds(Ab2 + k0, ldA), load_b_glb(BT2, K, n0, k0), acc);
  }
  int cn = n0 + (lane & 15);
  int rb = (lane >> 4) * 8;
#pragma unroll
  for (int j = 0; j < 8; ++j) {
    float v = acc[j] + (bias ? bias[cn] : 0.f);
    if (relu_act) v = fmaxf(v, 0.f);
    C[(size_t)(r0 + rb + j) * ldc + cn] = v;
  }
}

// ---------------------------------------- scores + softmax + weighted V sum
// s[m] = emb[conn[b,m,keycol]] . w[b] ; a = softmax(s) ; c[b] = sum_m a_m V[b,m]
// The 32KB V[b] block is copied global->LDS with GLOBAL_LOAD_ASYNC_TO_LDS_B128,
// fully overlapped with the score/softmax phase; completion via ASYNCcnt.
__global__ __launch_bounds__(256) void k_attn_core(
    const float* __restrict__ Wv, const int* __restrict__ conn, int keycol,
    const float* __restrict__ emb, const __bf16* __restrict__ Vbf,
    float* __restrict__ Cout) {
  __shared__ float wvec[256];
  __shared__ float sc[64];
  __shared__ __bf16 Vs[64 * 256];
  int b = blockIdx.x;

  // Kick off async copy of V[b] (64x256 bf16 = 32KB) into LDS.
  // Generic-pointer low 32 bits are the LDS byte address (flat aperture rule).
  {
    const char* gsrc = (const char*)(Vbf + (size_t)b * 64 * 256);
#pragma unroll
    for (int it = 0; it < 8; ++it) {
      unsigned byteoff = (unsigned)threadIdx.x * 16u + (unsigned)it * 4096u;
      unsigned ldsoff =
          (unsigned)(unsigned long long)(const void*)((const char*)Vs + byteoff);
      unsigned long long gaddr = (unsigned long long)(const void*)(gsrc + byteoff);
      asm volatile("global_load_async_to_lds_b128 %0, %1, off"
                   :: "v"(ldsoff), "v"(gaddr) : "memory");
    }
  }

  wvec[threadIdx.x] = Wv[(size_t)b * 256 + threadIdx.x];
  __syncthreads();

  int wv = threadIdx.x >> 5, lane = threadIdx.x & 31;
  for (int mm = 0; mm < 8; ++mm) {
    int m = wv * 8 + mm;
    int idx = conn[(size_t)(b * 64 + m) * 2 + keycol];
    const float* er = emb + (size_t)idx * 256;
    float p = 0.f;
#pragma unroll
    for (int t = 0; t < 8; ++t) {
      int d = lane + 32 * t;
      p = fmaf(er[d], wvec[d], p);
    }
#pragma unroll
    for (int off = 16; off; off >>= 1) p += __shfl_down(p, off);
    if (lane == 0) sc[m] = p;
  }
  __syncthreads();
  if (threadIdx.x < 32) {
    float s0 = sc[threadIdx.x], s1 = sc[threadIdx.x + 32];
    float mx = fmaxf(s0, s1);
#pragma unroll
    for (int off = 16; off; off >>= 1) mx = fmaxf(mx, __shfl_xor(mx, off));
    float e0 = __expf(s0 - mx), e1 = __expf(s1 - mx);
    float sm = e0 + e1;
#pragma unroll
    for (int off = 16; off; off >>= 1) sm += __shfl_xor(sm, off);
    float inv = 1.f / sm;
    sc[threadIdx.x] = e0 * inv;
    sc[threadIdx.x + 32] = e1 * inv;
  }
  // Each wave waits for its own async chunk, then barrier publishes all chunks.
  asm volatile("s_wait_asynccnt 0" ::: "memory");
  __syncthreads();

  int d = threadIdx.x;
  float acc = 0.f;
#pragma unroll 8
  for (int m = 0; m < 64; ++m) acc = fmaf(sc[m], (float)Vs[m * 256 + d], acc);
  Cout[(size_t)b * 256 + d] = acc;
}

// -------------------------------------------------- LayerNorm (x + e residual)
__global__ __launch_bounds__(256) void k_ln(
    const float* __restrict__ X, int ldx, const float* __restrict__ E, int lde,
    const float* __restrict__ gg, const float* __restrict__ bb,
    float* __restrict__ Out, int ldo, int W) {
  __shared__ float red[8];
  int b = blockIdx.x;
  int per = W >> 8;
  float v[2] = {0.f, 0.f};
  float s = 0.f;
  for (int i = 0; i < per; ++i) {
    int d = threadIdx.x + i * 256;
    v[i] = X[(size_t)b * ldx + d] + E[(size_t)b * lde + d];
    s += v[i];
  }
  float mean = block_sum256(s, red) / (float)W;
  float s2 = 0.f;
  for (int i = 0; i < per; ++i) { float t = v[i] - mean; s2 += t * t; }
  float var = block_sum256(s2, red) / (float)W;
  float rstd = rsqrtf(var + 1e-5f);
  for (int i = 0; i < per; ++i) {
    int d = threadIdx.x + i * 256;
    Out[(size_t)b * ldo + d] = (v[i] - mean) * rstd * gg[d] + bb[d];
  }
}

// ================================================================= launcher
extern "C" void kernel_launch(void* const* d_in, const int* in_sizes, int n_in,
                              void* d_out, int out_size, void* d_ws, size_t ws_size,
                              hipStream_t stream) {
  const int*   entity_pairs = (const int*)  d_in[0];
  const int*   left_conn    = (const int*)  d_in[1];
  const int*   right_conn   = (const int*)  d_in[3];
  const float* emb   = (const float*)d_in[5];
  const float* Wq_t  = (const float*)d_in[6];
  const float* Wk_t  = (const float*)d_in[7];
  const float* Wv_t  = (const float*)d_in[8];
  const float* Wb_t  = (const float*)d_in[9];
  const float* Wo_t  = (const float*)d_in[10];
  const float* bo_t  = (const float*)d_in[11];
  const float* Wq_e  = (const float*)d_in[12];
  const float* Wk_e  = (const float*)d_in[13];
  const float* Wv_e  = (const float*)d_in[14];
  const float* Wb_e  = (const float*)d_in[15];
  const float* Wo_e  = (const float*)d_in[16];
  const float* bo_e  = (const float*)d_in[17];
  const float* ln_g  = (const float*)d_in[18];
  const float* ln_b  = (const float*)d_in[19];
  const float* Wtail = (const float*)d_in[20];
  const float* Whead = (const float*)d_in[21];
  const float* Wrel  = (const float*)d_in[22];
  const float* W1    = (const float*)d_in[23];
  const float* b1    = (const float*)d_in[24];
  const float* W2    = (const float*)d_in[25];
  const float* ln1_g = (const float*)d_in[26];
  const float* ln1_b = (const float*)d_in[27];
  const float* Wg    = (const float*)d_in[28];
  const float* Ug    = (const float*)d_in[29];
  const float* bg    = (const float*)d_in[30];
  const float* Wr    = (const float*)d_in[31];
  const float* Ur    = (const float*)d_in[32];
  const float* br    = (const float*)d_in[33];
  (void)in_sizes; (void)n_in; (void)out_size; (void)ws_size;

  // --- workspace carve-out ---
  char* wp = (char*)d_ws;
  auto alloc = [&](size_t bytes) -> void* {
    void* p = (void*)wp;
    wp += (bytes + 255) & ~(size_t)255;
    return p;
  };
  __bf16* WrelB   = (__bf16*)alloc((size_t)512 * KREL * 2);
  __bf16* WgT     = (__bf16*)alloc(256 * 256 * 2);
  __bf16* UgT     = (__bf16*)alloc(256 * 256 * 2);
  __bf16* WrT     = (__bf16*)alloc(256 * 256 * 2);
  __bf16* UrT     = (__bf16*)alloc(256 * 256 * 2);
  __bf16* WtailB  = (__bf16*)alloc(256 * 256 * 2);
  __bf16* WheadB  = (__bf16*)alloc(256 * 256 * 2);
  __bf16* WscTB   = (__bf16*)alloc(256 * 256 * 2);
  __bf16* WscEB   = (__bf16*)alloc(256 * 256 * 2);
  __bf16* WvoTB   = (__bf16*)alloc(256 * 256 * 2);
  __bf16* WvoEB   = (__bf16*)alloc(256 * 256 * 2);
  __bf16* W1B     = (__bf16*)alloc(512 * 512 * 2);
  __bf16* W2B     = (__bf16*)alloc(512 * 512 * 2);
  float*  tmp1    = (float*) alloc(256 * 256 * 4);
  float*  tmp2    = (float*) alloc(256 * 256 * 4);
  float*  eL      = (float*) alloc((size_t)B_ * 256 * 4);
  float*  eR      = (float*) alloc((size_t)B_ * 256 * 4);
  __bf16* VheadB  = (__bf16*)alloc((size_t)B_ * M_ * 256 * 2);
  __bf16* VtailB  = (__bf16*)alloc((size_t)B_ * M_ * 256 * 2);
  float*  rel_emb = (float*) alloc((size_t)B_ * 512 * 4);
  float*  wbuf    = (float*) alloc((size_t)B_ * 256 * 4);
  float*  cbuf    = (float*) alloc((size_t)B_ * 256 * 4);
  float*  attbuf  = (float*) alloc((size_t)B_ * 256 * 4);
  float*  hbuf    = (float*) alloc((size_t)B_ * 512 * 4);
  float*  enh_ht  = (float*) alloc((size_t)B_ * 256 * 4);
  float*  enh_tt  = (float*) alloc((size_t)B_ * 256 * 4);
  float*  pair    = (float*) alloc((size_t)B_ * 512 * 4);
  float*  h1      = (float*) alloc((size_t)B_ * 512 * 4);
  float*  h2      = (float*) alloc((size_t)B_ * 512 * 4);

  auto cvt = [&](const float* src, __bf16* dst, int K, int N, int trans) {
    size_t tot = (size_t)K * N;
    k_cvt_bt<<<dim3((unsigned)((tot + 255) / 256)), 256, 0, stream>>>(src, dst, K, N, trans);
  };

  // 1. weight conversions
  cvt(Wrel,  WrelB, KREL, 512, 0);
  cvt(Wg,    WgT,   256, 256, 1);
  cvt(Ug,    UgT,   256, 256, 1);
  cvt(Wr,    WrT,   256, 256, 1);
  cvt(Ur,    UrT,   256, 256, 1);
  cvt(Wtail, WtailB,256, 256, 0);
  cvt(Whead, WheadB,256, 256, 0);
  cvt(W1,    W1B,   512, 512, 0);
  cvt(W2,    W2B,   512, 512, 0);

  // 2. precompute Wscore = Wk^T Wb Wq ; Wvo = Wo Wv (per module)
  k_mm256<<<256, 256, 0, stream>>>(Wb_t, Wq_t, tmp1, 0);
  k_mm256<<<256, 256, 0, stream>>>(Wk_t, tmp1, tmp2, 1);
  cvt(tmp2, WscTB, 256, 256, 0);
  k_mm256<<<256, 256, 0, stream>>>(Wo_t, Wv_t, tmp1, 0);
  cvt(tmp1, WvoTB, 256, 256, 0);
  k_mm256<<<256, 256, 0, stream>>>(Wb_e, Wq_e, tmp1, 0);
  k_mm256<<<256, 256, 0, stream>>>(Wk_e, tmp1, tmp2, 1);
  cvt(tmp2, WscEB, 256, 256, 0);
  k_mm256<<<256, 256, 0, stream>>>(Wo_e, Wv_e, tmp1, 0);
  cvt(tmp1, WvoEB, 256, 256, 0);

  // 3. gather pair embeddings
  k_gather_pair<<<B_, 256, 0, stream>>>(entity_pairs, emb, eL, eR);

  // 4. bilinear relation embedding (big WMMA GEMM, Wrel L2-resident)
  k_bilinear<<<dim3(B_ / 32, 4), 256, 0, stream>>>(eL, eR, WrelB, rel_emb);

  // 5. gated value matrices
  k_gates<<<(B_ * M_) / 16, 256, 0, stream>>>(left_conn,  emb, WgT, UgT, bg, VheadB);
  k_gates<<<(B_ * M_) / 16, 256, 0, stream>>>(right_conn, emb, WrT, UrT, br, VtailB);

  // attention + enhance block (algebraically refactored)
  auto attn_block = [&](const float* Q, int ldq, const int* conn, int keycol,
                        const __bf16* Vbf, const __bf16* Wsc, const __bf16* Wvo,
                        const float* bo, const float* eVec,
                        float* outLN, int ldo) {
    // w = Q @ Wscore^T
    k_gemm<<<dim3(B_ / 16, 2), 256, 0, stream>>>(
        Q, ldq, Wsc, nullptr, 0, nullptr, nullptr, 0, wbuf, 256, 256);
    // scores -> softmax -> c = a . V  (async V staging)
    k_attn_core<<<B_, 256, 0, stream>>>(wbuf, conn, keycol, emb, Vbf, cbuf);
    // att = c @ Wvo^T + bo
    k_gemm<<<dim3(B_ / 16, 2), 256, 0, stream>>>(
        cbuf, 256, Wvo, nullptr, 0, nullptr, bo, 0, attbuf, 256, 256);
    // h = relu(att @ Wtail^T + e @ Whead^T)
    k_gemm<<<dim3(B_ / 16, 2), 256, 0, stream>>>(
        attbuf, 256, WtailB, eVec, 256, WheadB, nullptr, 1, hbuf, 256, 256);
    // LN(h + e)
    k_ln<<<B_, 256, 0, stream>>>(hbuf, 256, eVec, 256, ln_g, ln_b, outLN, ldo, 256);
  };

  // 6. module t (relation path)
  attn_block(rel_emb + 0,   512, left_conn,  0, VheadB, WscTB, WvoTB, bo_t, eL, enh_ht, 256);
  attn_block(rel_emb + 256, 512, right_conn, 0, VtailB, WscTB, WvoTB, bo_t, eR, enh_tt, 256);
  // 7. module e (entity path) -> pair halves
  attn_block(enh_tt, 256, left_conn,  1, VheadB, WscEB, WvoEB, bo_e, eL, pair + 0,   512);
  attn_block(enh_ht, 256, right_conn, 1, VtailB, WscEB, WvoEB, bo_e, eR, pair + 256, 512);

  // 8. final MLP + LN
  k_gemm<<<dim3(B_ / 16, 4), 256, 0, stream>>>(
      pair, 512, W1B, nullptr, 0, nullptr, b1, 1, h1, 512, 512);
  k_gemm<<<dim3(B_ / 16, 4), 256, 0, stream>>>(
      h1, 512, W2B, nullptr, 0, nullptr, nullptr, 0, h2, 512, 512);
  k_ln<<<B_, 256, 0, stream>>>(h2, 512, pair, 512, ln1_g, ln1_b,
                               (float*)d_out, 512, 512);
}
